// Attention_11802570129581
// MI455X (gfx1250) — matrix operand order repaired and checked by reading, not compiled, not run on hardware
//
#include <hip/hip_runtime.h>

// ---------------------------------------------------------------------------
// MI455X (gfx1250) multi-head attention, f16 WMMA path, f32 accumulation.
// Compute-bound (~69 GFLOP vs ~2us of HBM time at 23.3 TB/s); all staging
// (~75 MB) is L2-resident (192 MB L2). GEMMs load per-lane-contiguous WMMA
// fragments straight from L2; flash-attention stages the (block-shared) K/V
// stream through LDS with the Tensor Data Mover (double-buffered,
// s_wait_tensorcnt-fenced), and computes softmax row-sums in the matrix pipe
// via a ones-matrix WMMA accumulator instead of lane reductions.
// ---------------------------------------------------------------------------

typedef __attribute__((ext_vector_type(16))) _Float16 v16h;
typedef __attribute__((ext_vector_type(8)))  _Float16 v8h;
typedef __attribute__((ext_vector_type(8)))  float    v8f;
typedef __attribute__((ext_vector_type(4)))  unsigned int u32x4;
typedef __attribute__((ext_vector_type(8)))  int          i32x8;
typedef __attribute__((ext_vector_type(4)))  int          i32x4;

#define DIM    1024
#define HEADS  16
#define HD     64
#define SEQ    2048
#define BATCH  2
#define MTOT   (BATCH * SEQ)   // 4096
#define N_QKV  (3 * DIM)       // 3072
#define SCALE  0.03125f        // DIM^-0.5 = 1/32  (exact in f16)

static __device__ __forceinline__ v16h cat8(v8h lo, v8h hi) {
  return __builtin_shufflevector(lo, hi, 0,1,2,3,4,5,6,7,8,9,10,11,12,13,14,15);
}

#define RFL(x) __builtin_amdgcn_readfirstlane((int)(x))

// Tensor Data Mover: 2-D (or 1-D if tile_d1==0) f16 tile, global -> LDS.
// D# built per CDNA5 ISA 8.3/8.4: group0 = {flags, lds_addr, global_addr,
// addr_hi|type=2}; group1 = dims/strides in 2-byte elements.
// This toolchain exposes the 6-arg builtin (g0, g1, g2, g3, g4, cpol).
static __device__ __forceinline__ void tdm_load_f16(
    unsigned lds_off, const void* gptr, unsigned tile_d0, unsigned tile_d1,
    unsigned tensor_d0, unsigned tensor_d1, unsigned stride0) {
  const unsigned long long ga = (unsigned long long)gptr;
  u32x4 g0;
  g0[0] = (unsigned)RFL(1u);                               // count=1, user D#
  g0[1] = (unsigned)RFL(lds_off);                          // lds_addr (bytes)
  g0[2] = (unsigned)RFL((unsigned)ga);                     // global_addr[31:0]
  g0[3] = (unsigned)RFL(((unsigned)(ga >> 32) & 0x01FFFFFFu) | (2u << 30));
  i32x8 g1;
  g1[0] = RFL(0x00010000);                                 // data_size = 2B
  g1[1] = RFL(tensor_d0 << 16);                            // tensor_dim0 lo
  g1[2] = RFL((tensor_d0 >> 16) | (tensor_d1 << 16));      // dim0 hi | dim1 lo
  g1[3] = RFL((tensor_d1 >> 16) | (tile_d0 << 16));        // dim1 hi | tile0
  g1[4] = RFL(tile_d1 & 0xffff);                           // tile1 (tile2=0)
  g1[5] = RFL(stride0);                                    // dim0_stride lo
  g1[6] = RFL(0);
  g1[7] = RFL(0);
  const i32x4 z4 = {0, 0, 0, 0};
  const i32x8 z8 = {0, 0, 0, 0, 0, 0, 0, 0};
  __builtin_amdgcn_tensor_load_to_lds(g0, g1, z4, z4, z8, 0);
}

// ---------------------------------------------------------------------------
// f32 -> f16 convert (x)
// ---------------------------------------------------------------------------
__global__ void convert_f16_kernel(const float* __restrict__ in,
                                   _Float16* __restrict__ out, int n) {
  for (int i = blockIdx.x * blockDim.x + threadIdx.x; i < n;
       i += gridDim.x * blockDim.x)
    out[i] = (_Float16)in[i];
}

// ---------------------------------------------------------------------------
// Transposing f32->f16 pack: w[K][Nn] -> wt[Nn][K]  (weights become the
// row-major "B^T" operand so WMMA B-fragments load 32 contiguous bytes/lane).
// ---------------------------------------------------------------------------
__global__ void transpose_pack_kernel(const float* __restrict__ w,
                                      _Float16* __restrict__ wt,
                                      int K, int Nn) {
  __shared__ float tile[32][33];
  const int n0 = blockIdx.x * 32, k0 = blockIdx.y * 32;
  tile[threadIdx.y][threadIdx.x] = w[(size_t)(k0 + threadIdx.y) * Nn + n0 + threadIdx.x];
  __syncthreads();
  wt[(size_t)(n0 + threadIdx.y) * K + k0 + threadIdx.x] =
      (_Float16)tile[threadIdx.x][threadIdx.y];
}

// ---------------------------------------------------------------------------
// GEMM: C[M][Nn] = A[M][K] * Bt[Nn][K]^T  via v_wmma_f32_16x16x32_f16.
// 256 threads = 8 waves; wave tile 32x64 (8 accumulators, 8 WMMAs / K-step).
// Workgroup tile 64x256.  F16OUT: store f16; else f32 + bias.
// ---------------------------------------------------------------------------
template <bool F16OUT>
__global__ __launch_bounds__(256) void gemm_wmma_kernel(
    const _Float16* __restrict__ A, const _Float16* __restrict__ Bt,
    void* __restrict__ Cout, const float* __restrict__ bias,
    int M, int Nn, int Kk) {
  const int lane = threadIdx.x & 31;
  const int wave = threadIdx.x >> 5;
  const int l16  = lane & 15;
  const int hlf  = lane >> 4;
  const int m_base = blockIdx.x * 64 + (wave >> 2) * 32;
  const int n_base = blockIdx.y * 256 + (wave & 3) * 64;

  const v8f vzero = {0.f, 0.f, 0.f, 0.f, 0.f, 0.f, 0.f, 0.f};
  v8f acc[2][4];
#pragma unroll
  for (int mi = 0; mi < 2; ++mi)
#pragma unroll
    for (int ni = 0; ni < 4; ++ni) acc[mi][ni] = vzero;

  for (int k0 = 0; k0 < Kk; k0 += 32) {
    // A fragments: 16x32, lane reads row l16, cols hlf*16..hlf*16+15
    v16h af[2];
#pragma unroll
    for (int mi = 0; mi < 2; ++mi) {
      const _Float16* ap = A + (size_t)(m_base + mi * 16 + l16) * Kk + k0;
      af[mi] = cat8(*(const v8h*)(ap + hlf * 16),
                    *(const v8h*)(ap + hlf * 16 + 8));
    }
    // B fragments: Bt row-major N x K, lane reads row l16, 32B at hlf*16
    v16h bf[4];
#pragma unroll
    for (int ni = 0; ni < 4; ++ni) {
      const _Float16* bp =
          Bt + (size_t)(n_base + ni * 16 + l16) * Kk + k0 + hlf * 16;
      bf[ni] = cat8(*(const v8h*)bp, *(const v8h*)(bp + 8));
    }
    // Prefetch next K-step (global_prefetch_b8); L2-resident thereafter.
    if (k0 + 32 < Kk) {
      __builtin_prefetch(A + (size_t)(m_base + l16) * Kk + k0 + 32, 0, 3);
      __builtin_prefetch(Bt + (size_t)(n_base + l16) * Kk + k0 + 32, 0, 3);
    }
#pragma unroll
    for (int mi = 0; mi < 2; ++mi)
#pragma unroll
      for (int ni = 0; ni < 4; ++ni)
        acc[mi][ni] = __builtin_amdgcn_wmma_f32_16x16x32_f16(
            false, af[mi], false, bf[ni], (short)0, acc[mi][ni], false, false);
  }

  // C layout: VGPR r, lane -> row = hlf*8 + r, col = l16 (per 16x16 tile)
#pragma unroll
  for (int mi = 0; mi < 2; ++mi)
#pragma unroll
    for (int ni = 0; ni < 4; ++ni)
#pragma unroll
      for (int r = 0; r < 8; ++r) {
        const int row = m_base + mi * 16 + hlf * 8 + r;
        const int col = n_base + ni * 16 + l16;
        const float v = acc[mi][ni][r];
        if (F16OUT)
          ((_Float16*)Cout)[(size_t)row * Nn + col] = (_Float16)v;
        else
          ((float*)Cout)[(size_t)row * Nn + col] = v + bias[col];
      }
}

// ---------------------------------------------------------------------------
// qkv[4096][3072] f16 -> Q (pre-scaled by DIM^-0.5), K [b][h][n][hd],
// V transposed [b][h][hd][n] (so P*V B-fragments are per-lane contiguous).
// ---------------------------------------------------------------------------
__global__ void split_heads_kernel(const _Float16* __restrict__ qkv,
                                   _Float16* __restrict__ Q,
                                   _Float16* __restrict__ K,
                                   _Float16* __restrict__ Vt) {
  const int total = MTOT * N_QKV;
  for (int i = blockIdx.x * blockDim.x + threadIdx.x; i < total;
       i += gridDim.x * blockDim.x) {
    const int m = i / N_QKV, c = i % N_QKV;
    const int which = c / DIM, cc = c % DIM;
    const int h = cc / HD, d = cc % HD;
    const int b = m / SEQ, t = m % SEQ;
    const _Float16 v = qkv[i];
    const size_t bh = (size_t)b * HEADS + h;
    if (which == 0)      Q[(bh * SEQ + t) * HD + d] = (_Float16)((float)v * SCALE);
    else if (which == 1) K[(bh * SEQ + t) * HD + d] = v;
    else                 Vt[(bh * HD + d) * SEQ + t] = v;
  }
}

// ---------------------------------------------------------------------------
// Flash attention. One wave per 16-row Q tile; 8 waves/block share one (b,h),
// so K/V tiles are TDM-copied into double-buffered LDS once per block.
// Per 32-key step: 4 WMMAs S=Q*K^T, 1 WMMA row-sum (ones matrix), 4 WMMAs
// O+=P*V. Online softmax; l is carried by the ones-column accumulator o[4].
// ---------------------------------------------------------------------------
__global__ __launch_bounds__(256) void flash_attn_kernel(
    const _Float16* __restrict__ Q, const _Float16* __restrict__ Kb,
    const _Float16* __restrict__ Vt, _Float16* __restrict__ Oout) {
  __shared__ __align__(16) _Float16 kbuf[2][32][64];   // 2 x 4 KB
  __shared__ __align__(16) _Float16 vbuf[2][64][32];   // 2 x 4 KB
  __shared__ __align__(16) _Float16 pst[8][16][32];    // 8 KB (per-wave)

  const int lane = threadIdx.x & 31;
  const int wave = threadIdx.x >> 5;
  const int l16  = lane & 15;
  const int hlf  = lane >> 4;

  const int gw = blockIdx.x * 8 + wave;        // 0..4095
  const int b  = gw >> 11;                     // / (HEADS*128)
  const int h  = (gw >> 7) & (HEADS - 1);
  const int q0 = (gw & 127) << 4;
  const size_t bh = (size_t)b * HEADS + h;

  const _Float16* Kbase = Kb + bh * SEQ * HD;
  const _Float16* Vbase = Vt + bh * HD * SEQ;

  // Q A-fragments (pre-scaled), loaded once, reused for all 64 key steps
  const _Float16* qrow = Q + (bh * SEQ + q0 + l16) * HD;
  v16h aq[2];
#pragma unroll
  for (int s = 0; s < 2; ++s)
    aq[s] = cat8(*(const v8h*)(qrow + s * 32 + hlf * 16),
                 *(const v8h*)(qrow + s * 32 + hlf * 16 + 8));

  const v8f vzero = {0.f, 0.f, 0.f, 0.f, 0.f, 0.f, 0.f, 0.f};
  const v16h vones = {(_Float16)1.f, (_Float16)1.f, (_Float16)1.f, (_Float16)1.f,
                      (_Float16)1.f, (_Float16)1.f, (_Float16)1.f, (_Float16)1.f,
                      (_Float16)1.f, (_Float16)1.f, (_Float16)1.f, (_Float16)1.f,
                      (_Float16)1.f, (_Float16)1.f, (_Float16)1.f, (_Float16)1.f};
  v8f o[5];                                    // o[4] = ones column = row sums
#pragma unroll
  for (int t = 0; t < 5; ++t) o[t] = vzero;
  float mrow[8];
#pragma unroll
  for (int r = 0; r < 8; ++r) mrow[r] = -3.0e38f;

  // Prologue: TDM-copy first K/V tiles into buffer 0 (wave 0 issues).
  if (wave == 0) {
    tdm_load_f16((unsigned)(size_t)&kbuf[0][0][0], Kbase, 32 * HD, 0,
                 32 * HD, 0, 32 * HD);                       // 1-D, 4 KB
    tdm_load_f16((unsigned)(size_t)&vbuf[0][0][0], Vbase, 32, HD,
                 SEQ, HD, SEQ);                              // 2-D, 64 rows
  }

  for (int it = 0; it < SEQ / 32; ++it) {
    const int kt = it * 32;
    const int cur = it & 1;
    if (wave == 0) {
      if (kt + 32 < SEQ) {
        tdm_load_f16((unsigned)(size_t)&kbuf[cur ^ 1][0][0],
                     Kbase + (size_t)(kt + 32) * HD, 32 * HD, 0,
                     32 * HD, 0, 32 * HD);
        tdm_load_f16((unsigned)(size_t)&vbuf[cur ^ 1][0][0],
                     Vbase + kt + 32, 32, HD, SEQ, HD, SEQ);
        __builtin_amdgcn_s_wait_tensorcnt(2);   // current pair complete
      } else {
        __builtin_amdgcn_s_wait_tensorcnt(0);
      }
    }
    __syncthreads();                            // buf[cur] valid for all waves

    // S = Q * K^T  (two 16x16 key sub-tiles, K-dim 64 in two 32-steps)
    v8f sacc[2];
    sacc[0] = vzero; sacc[1] = vzero;
#pragma unroll
    for (int j = 0; j < 2; ++j) {
      const _Float16* krow = &kbuf[cur][j * 16 + l16][0];
#pragma unroll
      for (int s = 0; s < 2; ++s) {
        v16h bk = cat8(*(const v8h*)(krow + s * 32 + hlf * 16),
                       *(const v8h*)(krow + s * 32 + hlf * 16 + 8));
        sacc[j] = __builtin_amdgcn_wmma_f32_16x16x32_f16(
            false, aq[s], false, bk, (short)0, sacc[j], false, false);
      }
    }

    // Online softmax max-update (row = hlf*8 + r across 16 lanes)
    float corr[8];
#pragma unroll
    for (int r = 0; r < 8; ++r) {
      float mx = fmaxf(sacc[0][r], sacc[1][r]);
#pragma unroll
      for (int off = 1; off < 16; off <<= 1)
        mx = fmaxf(mx, __shfl_xor(mx, off, 32));
      const float mnew = fmaxf(mrow[r], mx);
      corr[r] = __expf(mrow[r] - mnew);
      mrow[r] = mnew;
#pragma unroll
      for (int j = 0; j < 2; ++j)
        sacc[j][r] = __expf(sacc[j][r] - mnew);
    }
#pragma unroll
    for (int t = 0; t < 5; ++t)
#pragma unroll
      for (int r = 0; r < 8; ++r) o[t][r] *= corr[r];

    // P: C-layout -> LDS -> A-layout fragment (wave-private, no barrier)
#pragma unroll
    for (int j = 0; j < 2; ++j)
#pragma unroll
      for (int r = 0; r < 8; ++r)
        pst[wave][hlf * 8 + r][j * 16 + l16] = (_Float16)sacc[j][r];
    asm volatile("s_wait_dscnt 0" ::: "memory");
    v16h pf = cat8(*(const v8h*)&pst[wave][l16][hlf * 16],
                   *(const v8h*)&pst[wave][l16][hlf * 16 + 8]);

    // O += P * V ; o[4] += P * ones (row sums in the matrix pipe)
#pragma unroll
    for (int t = 0; t < 4; ++t) {
      const _Float16* vrow = &vbuf[cur][t * 16 + l16][hlf * 16];
      v16h bv = cat8(*(const v8h*)vrow, *(const v8h*)(vrow + 8));
      o[t] = __builtin_amdgcn_wmma_f32_16x16x32_f16(
          false, pf, false, bv, (short)0, o[t], false, false);
    }
    o[4] = __builtin_amdgcn_wmma_f32_16x16x32_f16(
        false, pf, false, vones, (short)0, o[4], false, false);

    __syncthreads();                            // all reads of buf done
  }

  // Epilogue: O / l, merge heads back into [b][n][DIM] f16
#pragma unroll
  for (int t = 0; t < 4; ++t)
#pragma unroll
    for (int r = 0; r < 8; ++r) {
      const int row = q0 + hlf * 8 + r;
      const float val = o[t][r] / o[4][r];
      Oout[((size_t)b * SEQ + row) * DIM + h * HD + t * 16 + l16] =
          (_Float16)val;
    }
}

// ---------------------------------------------------------------------------
// Launcher
// ---------------------------------------------------------------------------
extern "C" void kernel_launch(void* const* d_in, const int* in_sizes, int n_in,
                              void* d_out, int out_size, void* d_ws,
                              size_t ws_size, hipStream_t stream) {
  const float* x     = (const float*)d_in[0];  // [2,2048,1024]
  const float* w_qkv = (const float*)d_in[1];  // [1024,3072]
  const float* w_out = (const float*)d_in[2];  // [1024,1024]
  const float* b_out = (const float*)d_in[3];  // [1024]
  float* out = (float*)d_out;

  char* ws = (char*)d_ws;
  _Float16* x_h   = (_Float16*)(ws + 0);          //  8.0 MB
  _Float16* wqkvt = (_Float16*)(ws + 8388608);    //  6.0 MB
  _Float16* woutt = (_Float16*)(ws + 14680064);   //  2.0 MB
  _Float16* qkv_h = (_Float16*)(ws + 16777216);   // 24.0 MB
  _Float16* Qb    = (_Float16*)(ws + 41943040);   //  8.0 MB
  _Float16* Kbuf  = (_Float16*)(ws + 50331648);   //  8.0 MB
  _Float16* Vt    = (_Float16*)(ws + 58720256);   //  8.0 MB
  _Float16* attn  = (_Float16*)(ws + 67108864);   //  8.0 MB  (~75.5 MB total)

  convert_f16_kernel<<<1024, 256, 0, stream>>>(x, x_h, MTOT * DIM);
  transpose_pack_kernel<<<dim3(N_QKV / 32, DIM / 32), dim3(32, 32), 0, stream>>>(
      w_qkv, wqkvt, DIM, N_QKV);
  transpose_pack_kernel<<<dim3(DIM / 32, DIM / 32), dim3(32, 32), 0, stream>>>(
      w_out, woutt, DIM, DIM);

  // qkv = x @ w_qkv  (M=4096, N=3072, K=1024)
  gemm_wmma_kernel<true><<<dim3(MTOT / 64, N_QKV / 256), 256, 0, stream>>>(
      x_h, wqkvt, qkv_h, nullptr, MTOT, N_QKV, DIM);

  split_heads_kernel<<<2048, 256, 0, stream>>>(qkv_h, Qb, Kbuf, Vt);

  // softmax(QK^T * scale) V  — 4096 waves, 8 waves/block, TDM-fed K/V
  flash_attn_kernel<<<512, 256, 0, stream>>>(Qb, Kbuf, Vt, attn);

  // out = attn @ w_out + b_out  (M=4096, N=1024, K=1024), fp32 output
  gemm_wmma_kernel<false><<<dim3(MTOT / 64, DIM / 256), 256, 0, stream>>>(
      attn, woutt, out, b_out, MTOT, DIM, DIM);
}
